// HetAgg_33775622816413
// MI455X (gfx1250) — matrix operand order repaired and checked by reading, not compiled, not run on hardware
//
#include <hip/hip_runtime.h>

typedef __bf16 bf16;
typedef __attribute__((ext_vector_type(16))) __bf16 v16bf;
typedef __attribute__((ext_vector_type(8)))  float  v8f;

union Frag {
  v16bf v;
  uint4 q[2];
};

__device__ __forceinline__ v8f wmma_bf16(const Frag& a, const Frag& b, v8f c) {
  return __builtin_amdgcn_wmma_f32_16x16x32_bf16(
      /*neg_a=*/false, a.v, /*neg_b=*/false, b.v,
      /*c_mod=*/(short)0, c, /*reuse_a=*/false, /*reuse_b=*/false);
}

// A fragment (16x32 bf16, row-major source, row stride ld elements).
// lane<16 : row=lane,     K = kb+0..7  and kb+16..23
// lane>=16: row=lane-16,  K = kb+8..15 and kb+24..31
__device__ __forceinline__ Frag loadA(const bf16* base, long ld, int kb, int mn, int half) {
  Frag f;
  const bf16* p = base + (long)mn * ld + kb + half * 8;
  f.q[0] = *(const uint4*)p;
  f.q[1] = *(const uint4*)(p + 16);
  return f;
}

// B fragment (32x16 bf16) from transposed weight Wt[N=128][K=ldk].
// lane<16 : col=n0+lane,     K = kb+0..7  and kb+16..23
// lane>=16: col=n0+lane-16,  K = kb+8..15 and kb+24..31
__device__ __forceinline__ Frag loadB(const bf16* Wt, long ldk, int n0, int kb, int mn, int half) {
  Frag f;
  const bf16* p = Wt + (long)(n0 + mn) * ldk + kb + half * 8;
  f.q[0] = *(const uint4*)p;
  f.q[1] = *(const uint4*)(p + 16);
  return f;
}

// ---------------- weight convert: in [batch][K][N=128] f32 -> out [batch][N][K] bf16
__global__ void k_transpose_bf16(const float* __restrict__ in, bf16* __restrict__ out,
                                 int K, int N) {
  const size_t bz = blockIdx.z;
  const float* src = in + bz * (size_t)K * N;
  bf16* dst = out + bz * (size_t)K * N;
  int x = blockIdx.x * blockDim.x + threadIdx.x;
  if (x < K * N) {
    int k = x / N;
    int n = x - k * N;
    dst[(size_t)n * K + k] = (bf16)src[x];
  }
}

// ---------------- cell gather: out[r,0:128] = bf16(tab[idx[r],0:128]); 32 threads/row
__global__ void k_gather_cell(const float* __restrict__ tab, const int* __restrict__ idx,
                              bf16* __restrict__ outb) {
  int t = blockIdx.x * blockDim.x + threadIdx.x;
  int r = t >> 5;
  int c = (t & 31) * 4;
  float4 f = *(const float4*)(tab + (size_t)idx[r] * 128 + c);
  union { bf16 h[4]; uint2 u; } pk;
  pk.h[0] = (bf16)f.x; pk.h[1] = (bf16)f.y; pk.h[2] = (bf16)f.z; pk.h[3] = (bf16)f.w;
  *(uint2*)(outb + (size_t)r * 128 + c) = pk.u;
}

// ---------------- fused gather + projection GEMM via WMMA
// Block: 128 threads (4 waves). Block computes 16 rows x 128 cols.
// out[r, n] = feats[ids[r], :] @ W[:, n] + bias[n], Wt is [128][F] bf16 (transposed W).
template <bool F32OUT>
__global__ void k_proj(const float* __restrict__ feats, int F,
                       const int* __restrict__ ids,
                       const bf16* __restrict__ Wt,
                       const float* __restrict__ bias,
                       void* __restrict__ outp) {
  __shared__ bf16 atile[16 * 64];

  const int tid  = threadIdx.x;
  const int lane = tid & 31;
  const int wave = tid >> 5;
  const int half = lane >> 4;
  const int mn   = lane & 15;
  const int r0   = blockIdx.x * 16;
  const int n0   = wave * 32;

  // staging assignment: thread -> (row = tid/8, 8 consecutive K elements)
  const int srow = tid >> 3;
  const int skof = (tid & 7) * 8;
  const long sid = ids[r0 + srow];
  const float* src = feats + sid * (long)F + skof;

  v8f acc0 = {};
  v8f acc1 = {};

  for (int kb = 0; kb < F; kb += 64) {
    if (kb + 64 < F) __builtin_prefetch(src + kb + 64, 0, 3);  // global_prefetch_b8
    // stage 16x64 f32 chunk -> bf16 LDS tile
    float4 fa = *(const float4*)(src + kb);
    float4 fb = *(const float4*)(src + kb + 4);
    union { bf16 h[8]; uint4 q; } pk;
    pk.h[0] = (bf16)fa.x; pk.h[1] = (bf16)fa.y; pk.h[2] = (bf16)fa.z; pk.h[3] = (bf16)fa.w;
    pk.h[4] = (bf16)fb.x; pk.h[5] = (bf16)fb.y; pk.h[6] = (bf16)fb.z; pk.h[7] = (bf16)fb.w;
    *(uint4*)&atile[srow * 64 + skof] = pk.q;
    __syncthreads();

#pragma unroll
    for (int kk = 0; kk < 64; kk += 32) {
      Frag a  = loadA(atile, 64, kk, mn, half);
      Frag b0 = loadB(Wt, F, n0,      kb + kk, mn, half);
      Frag b1 = loadB(Wt, F, n0 + 16, kb + kk, mn, half);
      acc0 = wmma_bf16(a, b0, acc0);
      acc1 = wmma_bf16(a, b1, acc1);
    }
    __syncthreads();
  }

  const float bias0 = bias[n0 + mn];
  const float bias1 = bias[n0 + 16 + mn];
#pragma unroll
  for (int v = 0; v < 8; v++) {
    int m = v + half * 8;
    float v0 = acc0[v] + bias0;
    float v1 = acc1[v] + bias1;
    size_t o0 = (size_t)(r0 + m) * 128 + n0 + mn;
    size_t o1 = (size_t)(r0 + m) * 128 + n0 + 16 + mn;
    if (F32OUT) {
      ((float*)outp)[o0] = v0;
      ((float*)outp)[o1] = v1;
    } else {
      ((bf16*)outp)[o0] = (bf16)v0;
      ((bf16*)outp)[o1] = (bf16)v1;
    }
  }
}

// ---------------- tanh RNN over S sampled neighbors.
// Block: 256 threads = 8 waves; block computes 16 batch rows x 128 cols;
// each wave owns ONE 16-col N-tile so that the loop-invariant weight
// fragments (4 k-steps x {Wx,Wh} = 64 VGPRs) stay resident without spills.
// h_s = tanh(x_s @ Wx + h_{s-1} @ Wh + b); h tile (16x128 bf16) lives in LDS.
// grid.y = neighbor type t (0=cell,1=drug,2=gene).
__global__ void k_rnn(const bf16* __restrict__ x_cell,
                      const bf16* __restrict__ x_drug,
                      const bf16* __restrict__ x_gene,
                      const bf16* __restrict__ WxT,  // [3][128][128] (this layer)
                      const bf16* __restrict__ WhT,  // [3][128][128]
                      const float* __restrict__ bvec,  // [3][128]
                      float* __restrict__ agg,         // [3][B][128]
                      int S, int B) {
  const int t = blockIdx.y;
  const bf16* x  = (t == 0) ? x_cell : (t == 1) ? x_drug : x_gene;
  const bf16* Wx = WxT + (long)t * 128 * 128;
  const bf16* Wh = WhT + (long)t * 128 * 128;
  const float* bb = bvec + t * 128;
  float* outp = agg + (long)t * B * 128;

  __shared__ bf16 htile[16 * 128];

  const int tid  = threadIdx.x;
  const int lane = tid & 31;
  const int wave = tid >> 5;
  const int half = lane >> 4;
  const int mn   = lane & 15;
  const int b0   = blockIdx.x * 16;
  const int n0   = wave * 16;

  // preload loop-invariant weight B fragments (registers, reused all S steps)
  Frag fwx[4], fwh[4];
#pragma unroll
  for (int kk = 0; kk < 4; kk++) {
    fwx[kk] = loadB(Wx, 128, n0, kk * 32, mn, half);
    fwh[kk] = loadB(Wh, 128, n0, kk * 32, mn, half);
  }

  // h0 = 0 (2048 bf16 / 256 threads = 8 each = one 16B store)
  uint4 z = {0, 0, 0, 0};
  *(uint4*)&htile[tid * 8] = z;
  __syncthreads();

  const float bias = bb[n0 + mn];
  float vals[8];

  for (int s = 0; s < S; s++) {
    v8f acc = {};
    const bf16* xbase = x + ((long)b0 * S + s) * 128;  // row m at + m*S*128
#pragma unroll
    for (int kk = 0; kk < 4; kk++) {
      Frag ax = loadA(xbase, (long)S * 128, kk * 32, mn, half);
      Frag ah = loadA(htile, 128, kk * 32, mn, half);
      acc = wmma_bf16(ax, fwx[kk], acc);
      acc = wmma_bf16(ah, fwh[kk], acc);
    }
#pragma unroll
    for (int v = 0; v < 8; v++) vals[v] = tanhf(acc[v] + bias);
    __syncthreads();  // all waves done reading old h
#pragma unroll
    for (int v = 0; v < 8; v++) {
      int m = v + half * 8;
      htile[m * 128 + n0 + mn] = (bf16)vals[v];
    }
    __syncthreads();
  }

#pragma unroll
  for (int v = 0; v < 8; v++) {
    int m = v + half * 8;
    outp[(size_t)(b0 + m) * 128 + n0 + mn] = vals[v];
  }
}

// ---------------- attention combine: one wave per batch row, 4 candidates.
__device__ __forceinline__ float dot4(float4 a, float4 b) {
  return a.x * b.x + a.y * b.y + a.z * b.z + a.w * b.w;
}

__global__ void k_attn(const float* h,           // [B][128]
                       const float* __restrict__ agg,  // [3][B][128]
                       const float* __restrict__ w,    // [256] (this layer)
                       float* out, int B) {
  const int tid  = threadIdx.x;
  const int lane = tid & 31;
  const int wave = tid >> 5;
  const int b = blockIdx.x * 4 + wave;
  const int d = lane * 4;

  float4 hv = *(const float4*)(h + (size_t)b * 128 + d);
  float4 c1 = *(const float4*)(agg + (size_t)b * 128 + d);
  float4 c2 = *(const float4*)(agg + ((size_t)B + b) * 128 + d);
  float4 c3 = *(const float4*)(agg + ((size_t)2 * B + b) * 128 + d);
  float4 wl = *(const float4*)(w + d);
  float4 wh = *(const float4*)(w + 128 + d);

  float dl = dot4(hv, wl);
  float d0 = dot4(hv, wh);
  float d1 = dot4(c1, wh);
  float d2 = dot4(c2, wh);
  float d3 = dot4(c3, wh);
#pragma unroll
  for (int o = 16; o > 0; o >>= 1) {
    dl += __shfl_xor(dl, o, 32);
    d0 += __shfl_xor(d0, o, 32);
    d1 += __shfl_xor(d1, o, 32);
    d2 += __shfl_xor(d2, o, 32);
    d3 += __shfl_xor(d3, o, 32);
  }
  auto lrelu = [](float x) { return x > 0.0f ? x : 0.01f * x; };
  float s0 = lrelu(dl + d0);
  float s1 = lrelu(dl + d1);
  float s2 = lrelu(dl + d2);
  float s3 = lrelu(dl + d3);
  float mx = fmaxf(fmaxf(s0, s1), fmaxf(s2, s3));
  float e0 = expf(s0 - mx), e1 = expf(s1 - mx), e2 = expf(s2 - mx), e3 = expf(s3 - mx);
  float inv = 1.0f / (e0 + e1 + e2 + e3);
  float a0 = e0 * inv, a1 = e1 * inv, a2 = e2 * inv, a3 = e3 * inv;

  float4 o;
  o.x = a0 * hv.x + a1 * c1.x + a2 * c2.x + a3 * c3.x;
  o.y = a0 * hv.y + a1 * c1.y + a2 * c2.y + a3 * c3.y;
  o.z = a0 * hv.z + a1 * c1.z + a2 * c2.z + a3 * c3.z;
  o.w = a0 * hv.w + a1 * c1.w + a2 * c2.w + a3 * c3.w;
  *(float4*)(out + (size_t)b * 128 + d) = o;
}

extern "C" void kernel_launch(void* const* d_in, const int* in_sizes, int n_in,
                              void* d_out, int out_size, void* d_ws, size_t ws_size,
                              hipStream_t stream) {
  (void)in_sizes; (void)n_in; (void)out_size; (void)ws_size;
  constexpr int B = 4096, S = 10, Lyr = 2;
  constexpr int R = B * S;  // 40960 neighbor rows per (layer,type)

  const float* drug_feats = (const float*)d_in[0];   // [4000][1024]
  const float* gene_feats = (const float*)d_in[1];   // [20000][2048]
  const float* cell_tab   = (const float*)d_in[2];   // [2000][128]
  const float* W_drug = (const float*)d_in[3];       // [1024][128]
  const float* b_drug = (const float*)d_in[4];       // [128]
  const float* W_gene = (const float*)d_in[5];       // [2048][128]
  const float* b_gene = (const float*)d_in[6];       // [128]
  const float* rnn_Wx = (const float*)d_in[7];       // [2][3][128][128]
  const float* rnn_Wh = (const float*)d_in[8];       // [2][3][128][128]
  const float* rnn_b  = (const float*)d_in[9];       // [2][3][128]
  const float* att_w  = (const float*)d_in[10];      // [2][256]
  const int* center_ids = (const int*)d_in[11];      // [B]
  const int* cell_neigh = (const int*)d_in[12];      // [2][B][S]
  const int* drug_neigh = (const int*)d_in[13];
  const int* gene_neigh = (const int*)d_in[14];
  float* out = (float*)d_out;                        // [B][128]

  // workspace carve (256B aligned)
  char* ws = (char*)d_ws;
  size_t off = 0;
  auto carve = [&](size_t bytes) -> void* {
    void* p = ws + off;
    off = (off + bytes + 255) & ~(size_t)255;
    return p;
  };
  bf16*  WtD   = (bf16*)carve((size_t)1024 * 128 * 2);   // W_drug^T bf16
  bf16*  WtG   = (bf16*)carve((size_t)2048 * 128 * 2);   // W_gene^T bf16
  bf16*  WxT   = (bf16*)carve((size_t)6 * 128 * 128 * 2);
  bf16*  WhT   = (bf16*)carve((size_t)6 * 128 * 128 * 2);
  float* hbuf  = (float*)carve((size_t)B * 128 * 4);
  bf16*  xcell = (bf16*)carve((size_t)R * 128 * 2);
  bf16*  xdrug = (bf16*)carve((size_t)R * 128 * 2);
  bf16*  xgene = (bf16*)carve((size_t)R * 128 * 2);
  float* agg   = (float*)carve((size_t)3 * B * 128 * 4);

  // one-time weight conversion (bf16, N-major for contiguous B fragments)
  k_transpose_bf16<<<dim3(512, 1, 1),  256, 0, stream>>>(W_drug, WtD, 1024, 128);
  k_transpose_bf16<<<dim3(1024, 1, 1), 256, 0, stream>>>(W_gene, WtG, 2048, 128);
  k_transpose_bf16<<<dim3(64, 1, 6),   256, 0, stream>>>(rnn_Wx, WxT, 128, 128);
  k_transpose_bf16<<<dim3(64, 1, 6),   256, 0, stream>>>(rnn_Wh, WhT, 128, 128);

  // center embeds: h = drug_feats[center_ids] @ W_drug + b  (f32)
  k_proj<true><<<dim3(B / 16), 128, 0, stream>>>(drug_feats, 1024, center_ids, WtD, b_drug, hbuf);

  for (int l = 0; l < Lyr; l++) {
    k_gather_cell<<<dim3((R * 32) / 256), 256, 0, stream>>>(cell_tab, cell_neigh + (size_t)l * R, xcell);
    k_proj<false><<<dim3(R / 16), 128, 0, stream>>>(drug_feats, 1024, drug_neigh + (size_t)l * R, WtD, b_drug, xdrug);
    k_proj<false><<<dim3(R / 16), 128, 0, stream>>>(gene_feats, 2048, gene_neigh + (size_t)l * R, WtG, b_gene, xgene);
    k_rnn<<<dim3(B / 16, 3), 256, 0, stream>>>(xcell, xdrug, xgene,
                                               WxT + (size_t)l * 3 * 128 * 128,
                                               WhT + (size_t)l * 3 * 128 * 128,
                                               rnn_b + (size_t)l * 3 * 128,
                                               agg, S, B);
    k_attn<<<dim3(B / 4), 128, 0, stream>>>(hbuf, agg, att_w + (size_t)l * 256,
                                            (l == Lyr - 1) ? out : hbuf, B);
  }
}
